// SinkhornRouter_2302102471518
// MI455X (gfx1250) — matrix operand order repaired and checked by reading, not compiled, not run on hardware
//
#include <hip/hip_runtime.h>

// CDNA5 / gfx1250, wave32. Fused Sinkhorn-router:
//  Prepass: split W (64x2048 fp32) into f16 hi/lo pair in ws (removes all B-side
//           conversion VALU from the GEMM hot loop; W stays L2-resident).
//  Phase 1: logits tile (128 tok x 64 exp) via split-f16 WMMA (fp32-class accuracy)
//  Phase 2: persistent grid-wide Sinkhorn iterations (cost tile resident in LDS)
//  Phase 3: top-2 + softmax-gather, written directly to d_out.
// d_out layout: [ scores f32 (16384*2) | indices i32 (16384*2) ] (tuple concat).

typedef __attribute__((ext_vector_type(16))) _Float16 v16h;
typedef __attribute__((ext_vector_type(8)))  float    v8f;

#define HID   2048
#define NEXP  64
#define NTOK  16384
#define TOKPB 128
#define NBLK  (NTOK / TOKPB)   // 128 blocks (one LDS-resident tile each)
#define TPB   256              // 8 wave32 waves
#define STR   65               // LDS row stride (floats) -> conflict-free column walks
#define KTILE 32
#define NKC   (HID / KTILE)    // 64 k-chunks
#define TOLV  1e-4f
#define EPSV  1e-8f
#define MAXIT 500

__device__ __forceinline__ void gridBarrier(int* cnt, int* gen) {
  __syncthreads();
  if (threadIdx.x == 0) {
    __threadfence();
    int g = __hip_atomic_load(gen, __ATOMIC_ACQUIRE, __HIP_MEMORY_SCOPE_AGENT);
    int a = __hip_atomic_fetch_add(cnt, 1, __ATOMIC_ACQ_REL, __HIP_MEMORY_SCOPE_AGENT);
    if (a == NBLK - 1) {
      __hip_atomic_store(cnt, 0, __ATOMIC_RELAXED, __HIP_MEMORY_SCOPE_AGENT);
      __hip_atomic_fetch_add(gen, 1, __ATOMIC_ACQ_REL, __HIP_MEMORY_SCOPE_AGENT);
    } else {
      while (__hip_atomic_load(gen, __ATOMIC_ACQUIRE, __HIP_MEMORY_SCOPE_AGENT) == g)
        __builtin_amdgcn_s_sleep(1);
    }
  }
  __syncthreads();
}

__global__ void sinkhorn_router_init(int* bar) {
  bar[0] = 0;  // arrival count
  bar[1] = 0;  // generation
}

// Split W into f16 hi + f16 lo (residual): W = hi + lo to ~2^-22 relative.
__global__ __launch_bounds__(256) void w_split_kernel(
    const float* __restrict__ Wr, _Float16* __restrict__ whi, _Float16* __restrict__ wlo) {
  const int i = blockIdx.x * 256 + threadIdx.x;   // 0 .. NEXP*HID-1
  float f = Wr[i];
  _Float16 h = (_Float16)f;
  whi[i] = h;
  wlo[i] = (_Float16)(f - (float)h);
}

__global__ __launch_bounds__(TPB) void sinkhorn_router_kernel(
    const float* __restrict__ x,         // (16384, 2048)
    const _Float16* __restrict__ whi,    // (64, 2048) f16 hi
    const _Float16* __restrict__ wlo,    // (64, 2048) f16 lo
    float* __restrict__ out,             // scores | indices
    float* __restrict__ gpartial,        // (NBLK, NEXP) column partial sums
    float* __restrict__ gd1,             // (NEXP)
    float* __restrict__ gerr,            // (1)
    int*   __restrict__ bar)             // (2)
{
  __shared__ float cost[TOKPB * STR];   // exp(logits) tile, ~32.5 KB
  __shared__ float d0s[TOKPB];
  __shared__ float d1s[NEXP];
  __shared__ float d1prev[NEXP];        // used by block 0 only
  __shared__ float colpart[4 * NEXP];
  __shared__ float red[NEXP];
  __shared__ float s_err;

  const int tid   = threadIdx.x;
  const int blk   = blockIdx.x;
  const int wave  = tid >> 5;           // 0..7 -> row tile of 16 tokens
  const int lane  = tid & 31;
  const int lrow  = lane & 15;
  const int lhalf = lane >> 4;          // 0/1: which K-half this lane carries

  // ---------------- Phase 1: GEMM logits = x @ W^T via split-f16 WMMA ----------
  {
    const int rowbase = blk * TOKPB + wave * 16;
    const float* xrow = x + (size_t)(rowbase + lrow) * HID;
    v8f acc[4] = {};                     // 4 expert tiles of 16

    for (int kc = 0; kc < NKC; ++kc) {
      // A fragment (16x32 f16): lane<16 -> K {0..15}, lane>=16 -> K {16..31}
      const int klo = kc * KTILE + lhalf * 16;
      union { float4 q[4]; float f[16]; } ax;
      ax.q[0] = *(const float4*)(xrow + klo);
      ax.q[1] = *(const float4*)(xrow + klo + 4);
      ax.q[2] = *(const float4*)(xrow + klo + 8);
      ax.q[3] = *(const float4*)(xrow + klo + 12);
      v16h ah, al;
#pragma unroll
      for (int i = 0; i < 16; ++i) {
        _Float16 h = (_Float16)ax.f[i];
        ah[i] = h;
        al[i] = (_Float16)(ax.f[i] - (float)h);
      }
#pragma unroll
      for (int ct = 0; ct < 4; ++ct) {
        // B fragment (32x16 f16): lanes 0-15 hold K0..15 of expert N=lrow,
        // lanes 16-31 K16..31. Pre-split -> direct packed loads, no VALU.
        const size_t boff = (size_t)(ct * 16 + lrow) * HID + kc * KTILE + lhalf * 16;
        v16h bh = *(const v16h*)(whi + boff);   // 32B, aligned
        v16h bl = *(const v16h*)(wlo + boff);
        // fp32-class accuracy: hi*hi + lo*hi + hi*lo (lo*lo ~ 2^-22, negligible)
        acc[ct] = __builtin_amdgcn_wmma_f32_16x16x32_f16(false, ah, false, bh, (short)0, acc[ct], false, false);
        acc[ct] = __builtin_amdgcn_wmma_f32_16x16x32_f16(false, al, false, bh, (short)0, acc[ct], false, false);
        acc[ct] = __builtin_amdgcn_wmma_f32_16x16x32_f16(false, ah, false, bl, (short)0, acc[ct], false, false);
      }
    }
    // C layout: VGPR r -> M = r + 8*lhalf, N = lrow. Store cost = exp(logit) into LDS.
#pragma unroll
    for (int ct = 0; ct < 4; ++ct)
#pragma unroll
      for (int r = 0; r < 8; ++r) {
        int m = wave * 16 + r + 8 * lhalf;
        cost[m * STR + ct * 16 + lrow] = expf(acc[ct][r]);
      }
  }

  if (tid < NEXP) { d1s[tid] = 1.0f; d1prev[tid] = 1.0f; }
  __syncthreads();

  // ---------------- Phase 2: Sinkhorn iterations (grid-wide) -------------------
  int it = 0;
  while (true) {
    // Row pass: d0_i = (1/n0) / (sum_j d1_j * c_ij + eps)
    if (tid < TOKPB) {
      const float* cr = &cost[tid * STR];
      float s = 0.f;
#pragma unroll 8
      for (int j = 0; j < NEXP; ++j) s += d1s[j] * cr[j];
      d0s[tid] = (1.0f / (float)NTOK) / (s + EPSV);
    }
    __syncthreads();
    // Column partials: thread (j = tid&63, q = tid>>6) sums 32 rows (fixed order)
    {
      const int j = tid & 63, q = tid >> 6, r0 = q * 32;
      float s = 0.f;
#pragma unroll 8
      for (int rr = 0; rr < 32; ++rr) s += d0s[r0 + rr] * cost[(r0 + rr) * STR + j];
      colpart[q * NEXP + j] = s;
    }
    __syncthreads();
    if (tid < NEXP) {
      float s = colpart[tid] + colpart[NEXP + tid] + colpart[2 * NEXP + tid] + colpart[3 * NEXP + tid];
      __hip_atomic_store(&gpartial[blk * NEXP + tid], s, __ATOMIC_RELAXED, __HIP_MEMORY_SCOPE_AGENT);
      __threadfence();
    }
    gridBarrier(&bar[0], &bar[1]);

    if (blk == 0) {
      if (tid < NEXP) {
        float cs = 0.f;
        for (int b = 0; b < NBLK; ++b)  // fixed order -> deterministic
          cs += __hip_atomic_load(&gpartial[b * NEXP + tid], __ATOMIC_RELAXED, __HIP_MEMORY_SCOPE_AGENT);
        float d1n = (1.0f / (float)NEXP) / (cs + EPSV);
        red[tid] = fabsf(d1prev[tid] - d1n);
        d1prev[tid] = d1n;
        __hip_atomic_store(&gd1[tid], d1n, __ATOMIC_RELAXED, __HIP_MEMORY_SCOPE_AGENT);
        __threadfence();
      }
      __syncthreads();
      if (tid == 0) {
        float e = 0.f;
        for (int j = 0; j < NEXP; ++j) e += red[j];
        e *= (1.0f / (float)NEXP);
        __hip_atomic_store(gerr, e, __ATOMIC_RELAXED, __HIP_MEMORY_SCOPE_AGENT);
        __threadfence();
      }
    }
    gridBarrier(&bar[0], &bar[1]);

    if (tid < NEXP)
      d1s[tid] = __hip_atomic_load(&gd1[tid], __ATOMIC_RELAXED, __HIP_MEMORY_SCOPE_AGENT);
    if (tid == 0)
      s_err = __hip_atomic_load(gerr, __ATOMIC_ACQUIRE, __HIP_MEMORY_SCOPE_AGENT);
    __syncthreads();
    ++it;
    if (s_err <= TOLV || it >= MAXIT) break;   // uniform across all threads
  }

  // ---------------- Phase 3: top-2 + softmax gather ----------------------------
  // Rank within a row by d1_j * c_ij (d0_i > 0 is row-constant, preserves order);
  // softmax(logits)_j == c_ij / sum_k c_ik exactly.
  if (tid < TOKPB) {
    const float* cr = &cost[tid * STR];
    float m1 = -1.f, m2 = -1.f, c1 = 0.f, c2 = 0.f, sum = 0.f;
    int i1 = 0, i2 = 0;
    for (int j = 0; j < NEXP; ++j) {          // strict '>' => ties keep lower index
      float c = cr[j];
      float k = d1s[j] * c;
      sum += c;
      if (k > m1)      { m2 = m1; i2 = i1; c2 = c1; m1 = k; i1 = j; c1 = c; }
      else if (k > m2) { m2 = k;  i2 = j;  c2 = c; }
    }
    const int token = blk * TOKPB + tid;
    const float inv = 1.0f / sum;
    out[token * 2 + 0] = c1 * inv;
    out[token * 2 + 1] = c2 * inv;
    int* oidx = (int*)(out + 2 * NTOK);
    oidx[token * 2 + 0] = i1;
    oidx[token * 2 + 1] = i2;
  }
}

extern "C" void kernel_launch(void* const* d_in, const int* in_sizes, int n_in,
                              void* d_out, int out_size, void* d_ws, size_t ws_size,
                              hipStream_t stream) {
  (void)in_sizes; (void)n_in; (void)out_size; (void)ws_size;
  const float* x  = (const float*)d_in[0];   // (2048, 8, 2048) fp32
  const float* Wr = (const float*)d_in[1];   // (64, 2048) fp32
  float* out = (float*)d_out;

  // Workspace layout (bytes):
  //   [0,        32768)  gpartial : NBLK*NEXP f32
  //   [32768,    33024)  gd1      : NEXP f32
  //   [33024,    33028)  gerr     : 1 f32
  //   [33028,    33036)  bar      : 2 i32
  //   [40960,   303104)  whi      : NEXP*HID f16  (32B-aligned)
  //   [303104,  565248)  wlo      : NEXP*HID f16
  char* wsb = (char*)d_ws;
  float*    gpartial = (float*)(wsb);
  float*    gd1      = (float*)(wsb + 32768);
  float*    gerr     = (float*)(wsb + 33024);
  int*      bar      = (int*)  (wsb + 33028);
  _Float16* whi      = (_Float16*)(wsb + 40960);
  _Float16* wlo      = whi + (size_t)NEXP * HID;

  sinkhorn_router_init<<<1, 1, 0, stream>>>(bar);
  w_split_kernel<<<(NEXP * HID) / 256, 256, 0, stream>>>(Wr, whi, wlo);
  sinkhorn_router_kernel<<<NBLK, TPB, 0, stream>>>(x, whi, wlo, out, gpartial, gd1, gerr, bar);
}
